// SigLipAttentionTome_64398739636962
// MI455X (gfx1250) — matrix operand order repaired and checked by reading, not compiled, not run on hardware
//
#include <hip/hip_runtime.h>
#include <hip/hip_bf16.h>

// ---------------------------------------------------------------------------
// SigLip attention + ToMe size bias, CDNA5 (gfx1250) WMMA implementation.
// B=16, N=729, D=1152, H=16, HD=72.
// Data path: bf16 WMMA (f32 accum), TDM/async-LDS staging for shared A tiles.
// ---------------------------------------------------------------------------

typedef __attribute__((ext_vector_type(16))) __bf16 v16bf;
typedef __attribute__((ext_vector_type(8)))  __bf16 v8bf;
typedef __attribute__((ext_vector_type(8)))  float  v8f;

#define BB     16
#define NN     729
#define DD     1152
#define HH     16
#define HDIM   72
#define NPAD   736          // 46 * 16  (key/query tile padding)
#define HDPAD  96           // 3 * 32   (k-dim padding for QK^T)
#define HDPADT 80           // 5 * 16   (n-dim padding for AV output)
#define MM     (BB*NN)      // 11664 = 729 * 16 (exact multiple of 16)
#define ETILES (DD/16)      // 72
#define MTILES (MM/16)      // 729
#define KSTEPS_D (DD/32)    // 36
#define QT_PER_BH 46
#define SCALE_QK 0.11785113019775793f   // 72^-0.5

// ---------------------------------------------------------------------------
// gfx1250 async / TDM helpers
// ---------------------------------------------------------------------------
#if defined(__HIP_DEVICE_COMPILE__)
#if __has_builtin(__builtin_amdgcn_tensor_load_to_lds)
#define HAVE_TDM 1
#endif
#endif
#ifndef HAVE_TDM
#define HAVE_TDM 0
#endif

// One 16-byte lane-granular async copy global -> LDS (ASYNCcnt-tracked).
__device__ __forceinline__ void async_b128_to_lds(unsigned lds_addr, const void* g) {
  unsigned long long ga = (unsigned long long)(size_t)g;
  asm volatile("global_load_async_to_lds_b128 %0, %1, off"
               :: "v"(lds_addr), "v"(ga) : "memory");
}
__device__ __forceinline__ void wait_asynccnt0() {
  asm volatile("s_wait_asynccnt 0" ::: "memory");
}

#if HAVE_TDM
typedef unsigned int u32x4 __attribute__((ext_vector_type(4)));
typedef int          i32x4 __attribute__((ext_vector_type(4)));
typedef int          i32x8 __attribute__((ext_vector_type(8)));

// 1-D contiguous TDM load: nelem 2-byte elements, global -> LDS.
// D# packing per CDNA5 ISA (count=1, type=2,
// data_size=2B, tensor_dim0 = tile_dim0 = nelem, no padding, no multicast).
__device__ __forceinline__ void tdm_load_1d_to_lds(const void* gsrc, unsigned lds_addr,
                                                   unsigned nelem) {
  unsigned long long ga = (unsigned long long)(size_t)gsrc;
  u32x4 g0;
  g0.x = 1u;                                    // count=1 (valid), user mode
  g0.y = lds_addr;                              // lds_addr -> bits [63:32]
  g0.z = (unsigned)ga;                          // global_addr low
  g0.w = (unsigned)(ga >> 32) | 0x80000000u;    // global_addr high | type=2
  i32x8 g1;
  g1[0] = 0x00010000;                           // workgroup_mask=0, data_size=1 (2B)
  g1[1] = (int)((nelem & 0xFFFFu) << 16);       // tensor_dim0[15:0] -> bits 63:48
  g1[2] = (int)((nelem >> 16) | (1u << 16));    // tensor_dim0[31:16], tensor_dim1=1
  g1[3] = (int)((nelem & 0xFFFFu) << 16);       // tile_dim0 -> bits 127:112
  g1[4] = 0;                                    // tile_dim1=0, tile_dim2=0 (unused)
  g1[5] = (int)nelem;                           // tensor_dim0_stride low
  g1[6] = 0;
  g1[7] = 0;
  i32x4 z4 = {0, 0, 0, 0};
#if __clang_major__ >= 23
  i32x8 z8 = {0, 0, 0, 0, 0, 0, 0, 0};
  __builtin_amdgcn_tensor_load_to_lds(g0, g1, z4, z4, z8, 0);
#else
  __builtin_amdgcn_tensor_load_to_lds(g0, g1, z4, z4, 0);
#endif
}
#endif  // HAVE_TDM

// Stage a contiguous block of bf16 (nelem8 * 8 elements) into LDS, whole block
// cooperating. Uses TDM when available (wave 0 issues, TENSORcnt wait), else
// per-thread async-to-LDS b128 chunks (ASYNCcnt wait). Ends with barrier.
__device__ __forceinline__ void stage_to_lds(const __bf16* gsrc, __bf16* lds_dst,
                                             unsigned nchunks16B) {
#if HAVE_TDM
  if (threadIdx.x < 32u) {
    tdm_load_1d_to_lds(gsrc, (unsigned)(size_t)lds_dst, nchunks16B * 8u);
    __builtin_amdgcn_s_wait_tensorcnt((short)0);
  }
#else
  for (unsigned c = threadIdx.x; c < nchunks16B; c += blockDim.x)
    async_b128_to_lds((unsigned)(size_t)(lds_dst + c * 8u), gsrc + c * 8u);
  wait_asynccnt0();
#endif
  __syncthreads();
}

// ---------------------------------------------------------------------------
// fp32 -> bf16 conversion
// ---------------------------------------------------------------------------
__global__ __launch_bounds__(256) void cvt_f32_bf16(const float* __restrict__ src,
                                                    __bf16* __restrict__ dst, int n) {
  int i = blockIdx.x * blockDim.x + threadIdx.x;
  if (i < n) dst[i] = (__bf16)src[i];
}

// ---------------------------------------------------------------------------
// QKV projection: y = x @ W^T + b  (M=11664, K=1152, E=1152) x 3 matrices.
// Block = 8 waves sharing one 16-row A tile (staged once into LDS via TDM);
// each wave owns one 16x16 output tile, wmma_f32_16x16x32_bf16 over 36 k-steps.
// Q,K -> [B,H,NPAD,HDPAD] bf16; V -> transposed [B,H,HDPADT,NPAD] bf16.
// Grid is exact: 3*52488 tiles / 8 waves per block (divides evenly).
// ---------------------------------------------------------------------------
__global__ __launch_bounds__(256) void qkv_gemm(
    const __bf16* __restrict__ Xb,
    const __bf16* __restrict__ Wqb, const __bf16* __restrict__ Wkb,
    const __bf16* __restrict__ Wvb,
    const float* __restrict__ bq, const float* __restrict__ bk,
    const float* __restrict__ bv,
    __bf16* __restrict__ Qb, __bf16* __restrict__ Kb, __bf16* __restrict__ Vtb)
{
  __shared__ __attribute__((aligned(16))) __bf16 Xs[16 * DD];   // 36,864 B

  const unsigned lane = threadIdx.x & 31u;
  const unsigned wave = threadIdx.x >> 5;
  const unsigned half = lane >> 4;
  const unsigned l15  = lane & 15u;

  const unsigned TILES = MTILES * ETILES;           // 52488 (divisible by 8)
  const unsigned gw0 = blockIdx.x * 8u;             // block-uniform base tile
  const unsigned mat = gw0 / TILES;                 // 0=Q 1=K 2=V (uniform)
  const unsigned t0  = gw0 - mat * TILES;
  const unsigned mt  = t0 / ETILES;                 // uniform (72 % 8 == 0)
  const unsigned et  = (t0 - mt * ETILES) + wave;
  const unsigned m0 = mt * 16u, e0 = et * 16u;

  // stage shared A tile (16 rows x 1152 bf16, contiguous) into LDS
  stage_to_lds(Xb + (size_t)m0 * DD, Xs, (16u * DD) / 8u);

  const __bf16* Wb   = (mat == 0) ? Wqb : (mat == 1) ? Wkb : Wvb;
  const float*  bias = (mat == 0) ? bq  : (mat == 1) ? bk  : bv;
  const __bf16* arow = Xs + (size_t)l15 * DD;               // LDS
  const __bf16* brow = Wb + (size_t)(e0 + l15) * DD;        // global

  v8f c = {};
  for (unsigned ks = 0; ks < KSTEPS_D; ++ks) {
    const unsigned k0 = ks * 32u;
    // A fragment (16x32 bf16): lane = row, chunks at k0+8*half, k0+16+8*half
    v8bf a0 = *(const v8bf*)(arow + k0 + 8u * half);
    v8bf a1 = *(const v8bf*)(arow + k0 + 16u + 8u * half);
    // B fragment (32x16 bf16): lane = column e, chunks at k0+8*half, k0+16+8*half
    v8bf b0 = *(const v8bf*)(brow + k0 + 8u * half);
    v8bf b1 = *(const v8bf*)(brow + k0 + 16u + 8u * half);
    if (ks + 1 < KSTEPS_D) __builtin_prefetch(brow + k0 + 32u, 0, 1);
    v16bf a, b;
#pragma unroll
    for (int i = 0; i < 8; ++i) { a[i] = a0[i]; a[8 + i] = a1[i];
                                  b[i] = b0[i]; b[8 + i] = b1[i]; }
    c = __builtin_amdgcn_wmma_f32_16x16x32_bf16(false, a, false, b,
                                                (short)0, c, false, false);
  }

  const unsigned e  = e0 + l15;
  const float bval  = bias[e];
  const unsigned h  = e / HDIM;
  const unsigned hd = e - h * HDIM;

  if (mat == 2) {
#pragma unroll
    for (int v = 0; v < 8; ++v) {
      const unsigned m = m0 + (unsigned)v + 8u * half;
      const unsigned bidx = m / NN;
      const unsigned n = m - bidx * NN;
      Vtb[(size_t)((bidx * HH + h) * HDPADT + hd) * NPAD + n] = (__bf16)(c[v] + bval);
    }
  } else {
    __bf16* dst = (mat == 0) ? Qb : Kb;
#pragma unroll
    for (int v = 0; v < 8; ++v) {
      const unsigned m = m0 + (unsigned)v + 8u * half;
      const unsigned bidx = m / NN;
      const unsigned n = m - bidx * NN;
      dst[(size_t)((bidx * HH + h) * NPAD + n) * HDPAD + hd] = (__bf16)(c[v] + bval);
    }
  }
}

// ---------------------------------------------------------------------------
// Fused attention per (b,h,16-query tile):
//   Q tile staged once via global_load_async_to_lds (ASYNCcnt path)
//   scores = Q K^T * scale + log(size_k)   (WMMA, 8 waves over key tiles)
//   row softmax in LDS, stream fp32 attn to d_out (the 544 MB bound)
//   O = P @ V (WMMA, waves 0..4 own hd tiles), bf16 O1 for the out-proj
// ---------------------------------------------------------------------------
__global__ __launch_bounds__(256, 1) void attn_fused(
    const __bf16* __restrict__ Qb, const __bf16* __restrict__ Kb,
    const __bf16* __restrict__ Vtb, const float* __restrict__ size_in,
    float* __restrict__ attn_out, __bf16* __restrict__ O1)
{
  __shared__ float S[16][NPAD + 8];                            // 47,616 B
  __shared__ __attribute__((aligned(16))) __bf16 Qs[16 * HDPAD]; // 3,072 B
  __shared__ float red[16][16];
  __shared__ float rowred[16];

  const unsigned lane = threadIdx.x & 31u;
  const unsigned wave = threadIdx.x >> 5;
  const unsigned half = lane >> 4;
  const unsigned l15  = lane & 15u;

  const unsigned blk = blockIdx.x;
  const unsigned qt  = blk % QT_PER_BH;
  const unsigned bh  = blk / QT_PER_BH;       // b*H + h
  const unsigned b   = bh / HH;
  const unsigned h   = bh - b * HH;
  const unsigned q0  = qt * 16u;

  const __bf16* qbase = Qb + (size_t)bh * NPAD * HDPAD;
  const __bf16* kbase = Kb + (size_t)bh * NPAD * HDPAD;

  // ---- async-stage the Q tile (16 x 96 bf16 = 192 x 16B chunks) ----
  {
    const __bf16* qtile = qbase + (size_t)q0 * HDPAD;  // rows contiguous
    for (unsigned c = threadIdx.x; c < 192u; c += 256u)
      async_b128_to_lds((unsigned)(size_t)(Qs + c * 8u), qtile + c * 8u);
    wait_asynccnt0();
    __syncthreads();
  }

  // A fragments of the Q tile from LDS (reused across all key tiles)
  v16bf aQ[3];
#pragma unroll
  for (int ks = 0; ks < 3; ++ks) {
    v8bf a0 = *(const v8bf*)(Qs + l15 * HDPAD + ks * 32 + 8u * half);
    v8bf a1 = *(const v8bf*)(Qs + l15 * HDPAD + ks * 32 + 16u + 8u * half);
#pragma unroll
    for (int i = 0; i < 8; ++i) { aQ[ks][i] = a0[i]; aQ[ks][8 + i] = a1[i]; }
  }

  // ---- scores ----
  for (unsigned kt = wave; kt < QT_PER_BH; kt += 8u) {
    const unsigned key0 = kt * 16u;
    const __bf16* krow = kbase + (size_t)(key0 + l15) * HDPAD;
    v8f c = {};
#pragma unroll
    for (int ks = 0; ks < 3; ++ks) {
      v8bf b0 = *(const v8bf*)(krow + ks * 32 + 8u * half);
      v8bf b1 = *(const v8bf*)(krow + ks * 32 + 16u + 8u * half);
      v16bf bf;
#pragma unroll
      for (int i = 0; i < 8; ++i) { bf[i] = b0[i]; bf[8 + i] = b1[i]; }
      c = __builtin_amdgcn_wmma_f32_16x16x32_bf16(false, aQ[ks], false, bf,
                                                  (short)0, c, false, false);
    }
    const unsigned key = key0 + l15;           // column fixed per lane
    const bool kvalid = key < NN;
    const float ls = kvalid ? __logf(size_in[b * NN + key]) : 0.0f;
#pragma unroll
    for (int v = 0; v < 8; ++v) {
      const unsigned m = (unsigned)v + 8u * half;
      S[m][key] = kvalid ? (c[v] * SCALE_QK + ls) : -1e30f;
    }
  }
  __syncthreads();

  // ---- softmax (row r handled by 16 threads) ----
  {
    const unsigned r = threadIdx.x >> 4;
    const unsigned j = threadIdx.x & 15u;
    float mx = -1e30f;
    for (unsigned k = j; k < NPAD; k += 16u) mx = fmaxf(mx, S[r][k]);
    red[r][j] = mx;
    __syncthreads();
    if (j == 0) {
      float m2 = red[r][0];
#pragma unroll
      for (int i = 1; i < 16; ++i) m2 = fmaxf(m2, red[r][i]);
      rowred[r] = m2;
    }
    __syncthreads();
    const float rmax = rowred[r];
    float s = 0.0f;
    for (unsigned k = j; k < NPAD; k += 16u) {
      const float e = __expf(S[r][k] - rmax);
      S[r][k] = e; s += e;
    }
    red[r][j] = s;
    __syncthreads();
    if (j == 0) {
      float s2 = 0.0f;
#pragma unroll
      for (int i = 0; i < 16; ++i) s2 += red[r][i];
      rowred[r] = s2;
    }
    __syncthreads();
    const float inv = 1.0f / rowred[r];
    const unsigned q = q0 + r;
    float* arow = attn_out + ((size_t)bh * NN + q) * NN;
    for (unsigned k = j; k < NPAD; k += 16u) {
      const float p = S[r][k] * inv;
      S[r][k] = p;
      if (q < NN && k < NN) arow[k] = p;
    }
  }
  __syncthreads();

  // ---- O = P @ V : waves 0..4 each own one 16-wide hd tile ----
  if (wave < 5u) {
    const unsigned ht = wave;
    const __bf16* vrow = Vtb + (size_t)(bh * HDPADT + ht * 16u + l15) * NPAD;
    v8f c = {};
    for (unsigned ks = 0; ks < NPAD / 32u; ++ks) {       // 23 steps
      const unsigned k0 = ks * 32u;
      v16bf af, bf;
#pragma unroll
      for (int i = 0; i < 8; ++i) {
        af[i]     = (__bf16)S[l15][k0 + 8u * half + (unsigned)i];
        af[8 + i] = (__bf16)S[l15][k0 + 16u + 8u * half + (unsigned)i];
      }
      v8bf b0 = *(const v8bf*)(vrow + k0 + 8u * half);
      v8bf b1 = *(const v8bf*)(vrow + k0 + 16u + 8u * half);
#pragma unroll
      for (int i = 0; i < 8; ++i) { bf[i] = b0[i]; bf[8 + i] = b1[i]; }
      c = __builtin_amdgcn_wmma_f32_16x16x32_bf16(false, af, false, bf,
                                                  (short)0, c, false, false);
    }
    const unsigned hd = ht * 16u + l15;
    if (hd < HDIM) {
#pragma unroll
      for (int v = 0; v < 8; ++v) {
        const unsigned q = q0 + (unsigned)v + 8u * half;
        if (q < NN)
          O1[(size_t)(b * NN + q) * DD + h * HDIM + hd] = (__bf16)c[v];
      }
    }
  }
}

// ---------------------------------------------------------------------------
// metric = mean over heads of K  -> (B, N, HD) fp32
// ---------------------------------------------------------------------------
__global__ __launch_bounds__(256) void metric_kernel(const __bf16* __restrict__ Kb,
                                                     float* __restrict__ metric) {
  const unsigned TOT = BB * NN * HDIM;
  unsigned i = blockIdx.x * blockDim.x + threadIdx.x;
  if (i >= TOT) return;
  const unsigned hd = i % HDIM;
  const unsigned tmp = i / HDIM;
  const unsigned n = tmp % NN;
  const unsigned b = tmp / NN;
  float s = 0.0f;
#pragma unroll
  for (int h = 0; h < HH; ++h)
    s += (float)Kb[(size_t)((b * HH + (unsigned)h) * NPAD + n) * HDPAD + hd];
  metric[i] = s * (1.0f / HH);
}

// ---------------------------------------------------------------------------
// Output projection: out = O1 @ Wo^T + bo -> fp32 d_out[0 .. B*N*D)
// Same block structure as qkv_gemm (52488 tiles, divides by 8 exactly).
// ---------------------------------------------------------------------------
__global__ __launch_bounds__(256) void oproj_gemm(
    const __bf16* __restrict__ O1, const __bf16* __restrict__ Wob,
    const float* __restrict__ bo, float* __restrict__ out)
{
  __shared__ __attribute__((aligned(16))) __bf16 As[16 * DD];   // 36,864 B

  const unsigned lane = threadIdx.x & 31u;
  const unsigned wave = threadIdx.x >> 5;
  const unsigned half = lane >> 4;
  const unsigned l15  = lane & 15u;

  const unsigned gw0 = blockIdx.x * 8u;
  const unsigned mt  = gw0 / ETILES;                 // uniform per block
  const unsigned et  = (gw0 - mt * ETILES) + wave;
  const unsigned m0 = mt * 16u, e0 = et * 16u;

  stage_to_lds(O1 + (size_t)m0 * DD, As, (16u * DD) / 8u);

  const __bf16* arow = As  + (size_t)l15 * DD;
  const __bf16* brow = Wob + (size_t)(e0 + l15) * DD;

  v8f c = {};
  for (unsigned ks = 0; ks < KSTEPS_D; ++ks) {
    const unsigned k0 = ks * 32u;
    v8bf a0 = *(const v8bf*)(arow + k0 + 8u * half);
    v8bf a1 = *(const v8bf*)(arow + k0 + 16u + 8u * half);
    v8bf b0 = *(const v8bf*)(brow + k0 + 8u * half);
    v8bf b1 = *(const v8bf*)(brow + k0 + 16u + 8u * half);
    if (ks + 1 < KSTEPS_D) __builtin_prefetch(brow + k0 + 32u, 0, 1);
    v16bf a, b;
#pragma unroll
    for (int i = 0; i < 8; ++i) { a[i] = a0[i]; a[8 + i] = a1[i];
                                  b[i] = b0[i]; b[8 + i] = b1[i]; }
    c = __builtin_amdgcn_wmma_f32_16x16x32_bf16(false, a, false, b,
                                                (short)0, c, false, false);
  }
  const float bval = bo[e0 + l15];
#pragma unroll
  for (int v = 0; v < 8; ++v) {
    const unsigned m = m0 + (unsigned)v + 8u * half;
    out[(size_t)m * DD + e0 + l15] = c[v] + bval;
  }
}

// ---------------------------------------------------------------------------
// host launcher
// ---------------------------------------------------------------------------
static inline size_t alignup256(size_t x) { return (x + 255u) & ~(size_t)255u; }

extern "C" void kernel_launch(void* const* d_in, const int* in_sizes, int n_in,
                              void* d_out, int out_size, void* d_ws, size_t ws_size,
                              hipStream_t stream) {
  const float* hs      = (const float*)d_in[0];
  const float* size_in = (const float*)d_in[1];
  const float* Wq = (const float*)d_in[2];
  const float* bq = (const float*)d_in[3];
  const float* Wk = (const float*)d_in[4];
  const float* bk = (const float*)d_in[5];
  const float* Wv = (const float*)d_in[6];
  const float* bv = (const float*)d_in[7];
  const float* Wo = (const float*)d_in[8];
  const float* bo = (const float*)d_in[9];

  float* out    = (float*)d_out;
  float* attn   = out + (size_t)MM * DD;                  // B*H*N*N fp32
  float* metric = attn + (size_t)BB * HH * NN * NN;       // B*N*HD fp32

  // workspace carve-up (bf16 buffers)
  char* p = (char*)d_ws;
  __bf16* Xb  = (__bf16*)p; p += alignup256((size_t)MM * DD * 2);
  __bf16* Wqb = (__bf16*)p; p += alignup256((size_t)DD * DD * 2);
  __bf16* Wkb = (__bf16*)p; p += alignup256((size_t)DD * DD * 2);
  __bf16* Wvb = (__bf16*)p; p += alignup256((size_t)DD * DD * 2);
  __bf16* Wob = (__bf16*)p; p += alignup256((size_t)DD * DD * 2);
  const size_t qkElems = (size_t)BB * HH * NPAD * HDPAD;  // 18,087,936
  const size_t vtElems = (size_t)BB * HH * HDPADT * NPAD; // 15,073,280
  __bf16* Qb  = (__bf16*)p; p += alignup256(qkElems * 2);
  __bf16* Kb  = (__bf16*)p; p += alignup256(qkElems * 2);
  __bf16* Vtb = (__bf16*)p; p += alignup256(vtElems * 2);
  __bf16* O1  = (__bf16*)p; p += alignup256((size_t)MM * DD * 2);

  // zero padded Q/K/V so padding contributes exact zeros to WMMA
  hipMemsetAsync(Qb,  0, qkElems * 2, stream);
  hipMemsetAsync(Kb,  0, qkElems * 2, stream);
  hipMemsetAsync(Vtb, 0, vtElems * 2, stream);

  // fp32 -> bf16 conversions
  {
    const int nX = MM * DD, nW = DD * DD;
    cvt_f32_bf16<<<(nX + 255) / 256, 256, 0, stream>>>(hs, Xb, nX);
    cvt_f32_bf16<<<(nW + 255) / 256, 256, 0, stream>>>(Wq, Wqb, nW);
    cvt_f32_bf16<<<(nW + 255) / 256, 256, 0, stream>>>(Wk, Wkb, nW);
    cvt_f32_bf16<<<(nW + 255) / 256, 256, 0, stream>>>(Wv, Wvb, nW);
    cvt_f32_bf16<<<(nW + 255) / 256, 256, 0, stream>>>(Wo, Wob, nW);
  }

  // QKV projections (3 * 52488 wave-tiles, exactly 8 per block)
  qkv_gemm<<<(3u * MTILES * ETILES) / 8, 256, 0, stream>>>(
      Xb, Wqb, Wkb, Wvb, bq, bk, bv, Qb, Kb, Vtb);

  // fused scores + softmax + AV
  attn_fused<<<BB * HH * QT_PER_BH, 256, 0, stream>>>(
      Qb, Kb, Vtb, size_in, attn, O1);

  // metric (mean of K over heads)
  {
    const unsigned tot = BB * NN * HDIM;
    metric_kernel<<<(tot + 255) / 256, 256, 0, stream>>>(Kb, metric);
  }

  // output projection (52488 tiles, exactly 8 per block)
  oproj_gemm<<<(MTILES * ETILES) / 8, 256, 0, stream>>>(O1, Wob, bo, out);
}
